// PointTransformerCls_48112223650046
// MI455X (gfx1250) — compile-verified, matrix operand order checked by reading, not run on hardware
//
#include <hip/hip_runtime.h>

// ============================================================================
// PointTransformerCls forward for gfx1250 (MI455X, wave32, WMMA).
// All dense math -> v_wmma_f32_16x16x32_f16 (f32 storage, f16 LDS staging,
// f32 accumulate). Attention is flash-style (online softmax) so the
// B*H*2048*2048 score matrices (256MB each) are never materialized.
// GEMM staging: block-uniform fast/slow path selection (scalar branches,
// no EXEC juggling); fast path uses float4 (b128) global loads, edge path
// uses branchless clamped loads + cndmask.
// Workspace peak ~165MB (gfg set-abstraction grouped tensor dominates).
// ============================================================================

typedef __attribute__((ext_vector_type(16))) _Float16 v16h;
typedef __attribute__((ext_vector_type(8)))  float    v8f;

#define BB 4
#define NPTS 2048
#define PTS_ST 8   // padded row stride of pts (B, N, 8), cols 6..7 = 0

union AF { v16h v; unsigned int u[8]; };

__device__ __forceinline__ unsigned int pack2(float a, float b) {
  union { _Float16 h[2]; unsigned int u; } p;
  p.h[0] = (_Float16)a; p.h[1] = (_Float16)b; return p.u;
}

// ---------------------------------------------------------------------------
// Generic batched GEMM: C[z] = act(A[z](MxK) @ W(KxN) + bias), weights shared.
// Block tile 64x64, BK=32, 4 waves; per wave 16 rows x 64 cols (4 WMMA/step).
// A-frag lane map: m=l&15, kbase=(l>>4)*8, pair i -> k=(i>>2)*16+kbase+2*(i&3).
// B staged transposed Bsh[n][k] so frag pairs are dword LDS loads.
// ---------------------------------------------------------------------------
__global__ void __launch_bounds__(128) gemm_kernel(
    const float* __restrict__ A, const float* __restrict__ Wt,
    const float* __restrict__ bias, float* __restrict__ C,
    int M, int N, int K, int lda, long long sA, int ldc, long long sC, int act)
{
  __shared__ _Float16 Ash[64 * 34];
  __shared__ _Float16 Bsh[64 * 34];
  const int tid  = threadIdx.x;
  const int lane = tid & 31;
  const int w    = tid >> 5;
  const int bm   = blockIdx.y * 64;
  const int bn   = blockIdx.x * 64;
  const float* Ab = A + (long long)blockIdx.z * sA;
  float*       Cb = C + (long long)blockIdx.z * sC;

  v8f acc[4] = {};

  const int ar = tid >> 1;            // A row 0..63
  const int ah = (tid & 1) * 16;      // A k segment
  const int bk = tid >> 2;            // B k row 0..31
  const int bs = (tid & 3) * 16;      // B n segment
  const int kbase = (lane >> 4) * 8;
  const int mloc  = w * 16 + (lane & 15);
  // Block-uniform path selectors (scalar branches, no EXEC divergence).
  const bool ldaOK = (lda & 3) == 0;
  const bool nOK   = (N & 3) == 0;
  const bool fullM = (bm + 64 <= M);
  const bool fullN = (bn + 64 <= N);
  const int gm  = bm + ar;
  const int gmc = gm < M ? gm : (M - 1);

  for (int k0 = 0; k0 < K; k0 += 32) {
    const bool fullK = (k0 + 32 <= K);          // uniform
    // ---- stage A (64x32) -> Ash[m][k] ----
    if (fullM && ldaOK && fullK) {
      const float* src = Ab + (long long)gm * lda + k0 + ah;
      #pragma unroll
      for (int i = 0; i < 16; i += 4) {
        const float4 f = *(const float4*)(src + i);
        *(unsigned int*)&Ash[ar * 34 + ah + i]     = pack2(f.x, f.y);
        *(unsigned int*)&Ash[ar * 34 + ah + i + 2] = pack2(f.z, f.w);
      }
    } else {
      const bool rok = gm < M;
      const float* rowp = Ab + (long long)gmc * lda;
      #pragma unroll
      for (int i = 0; i < 16; i += 2) {
        const int ka  = k0 + ah + i;
        const int kb2 = ka + 1;
        float f0 = rowp[ka  < K ? ka  : (K - 1)];
        float f1 = rowp[kb2 < K ? kb2 : (K - 1)];
        f0 = (rok && ka  < K) ? f0 : 0.0f;
        f1 = (rok && kb2 < K) ? f1 : 0.0f;
        *(unsigned int*)&Ash[ar * 34 + ah + i] = pack2(f0, f1);
      }
    }
    // ---- stage B (32x64) -> Bsh[n][k] (transposed) ----
    if (fullN && nOK && fullK) {
      const float* src = Wt + (long long)(k0 + bk) * N + bn + bs;
      #pragma unroll
      for (int i = 0; i < 16; i += 4) {
        const float4 f = *(const float4*)(src + i);
        Bsh[(bs + i + 0) * 34 + bk] = (_Float16)f.x;
        Bsh[(bs + i + 1) * 34 + bk] = (_Float16)f.y;
        Bsh[(bs + i + 2) * 34 + bk] = (_Float16)f.z;
        Bsh[(bs + i + 3) * 34 + bk] = (_Float16)f.w;
      }
    } else {
      const int gk = k0 + bk;
      const bool kok = gk < K;
      const float* rowp = Wt + (long long)(kok ? gk : (K - 1)) * N;
      #pragma unroll
      for (int i = 0; i < 16; ++i) {
        const int gn = bn + bs + i;
        float f = rowp[gn < N ? gn : (N - 1)];
        f = (kok && gn < N) ? f : 0.0f;
        Bsh[(bs + i) * 34 + bk] = (_Float16)f;
      }
    }
    __syncthreads();
    AF a;
    #pragma unroll
    for (int i = 0; i < 8; ++i)
      a.u[i] = *(const unsigned int*)&Ash[mloc * 34 + ((i >> 2) * 16) + kbase + 2 * (i & 3)];
    #pragma unroll
    for (int t = 0; t < 4; ++t) {
      AF bf;
      const int nloc = t * 16 + (lane & 15);
      #pragma unroll
      for (int i = 0; i < 8; ++i)
        bf.u[i] = *(const unsigned int*)&Bsh[nloc * 34 + ((i >> 2) * 16) + kbase + 2 * (i & 3)];
      acc[t] = __builtin_amdgcn_wmma_f32_16x16x32_f16(false, a.v, false, bf.v,
                                                      (short)0, acc[t], false, false);
    }
    __syncthreads();
  }
  // ---- epilogue: bias + optional ReLU ----
  if (fullM && fullN) {
    #pragma unroll
    for (int t = 0; t < 4; ++t) {
      const int gn = bn + t * 16 + (lane & 15);
      const float bv = bias[gn];
      #pragma unroll
      for (int i = 0; i < 8; ++i) {
        const int gmm = bm + w * 16 + i + (lane >> 4) * 8;
        float v = acc[t][i] + bv;
        if (act) v = fmaxf(v, 0.0f);
        Cb[(long long)gmm * ldc + gn] = v;
      }
    }
  } else {
    #pragma unroll
    for (int t = 0; t < 4; ++t) {
      const int gn = bn + t * 16 + (lane & 15);
      #pragma unroll
      for (int i = 0; i < 8; ++i) {
        const int gmm = bm + w * 16 + i + (lane >> 4) * 8;
        if (gmm < M && gn < N) {
          float v = acc[t][i] + bias[gn];
          if (act) v = fmaxf(v, 0.0f);
          Cb[(long long)gmm * ldc + gn] = v;
        }
      }
    }
  }
}

// ---------------------------------------------------------------------------
// Flash attention (Lq%64==0, Lk%32==0). Q/K/V/O layout (B, L, H*DK) f32.
// One block = 4 waves = 64 q rows per (b,h); K/V 32-key tiles staged in LDS
// block-wide via float4 loads; P transposed to A-fragment layout via a
// per-wave LDS bounce. Online softmax in the WMMA C layout (shfl_xor w16).
// ---------------------------------------------------------------------------
template <int DK>
__global__ void __launch_bounds__(128) flash_attn_kernel(
    const float* __restrict__ Q, const float* __restrict__ Km,
    const float* __restrict__ Vm, float* __restrict__ O, int Lq, int Lk)
{
  constexpr int NQF = DK / 32;     // Q A-frags over contraction dim
  constexpr int NOF = DK / 16;     // O frags over dv (= DK)
  constexpr int KST = DK + 2;      // Ksh padded row stride (halfs)
  __shared__ _Float16 Ksh[32 * KST];   // [key 0..31][dk]
  __shared__ _Float16 Vsh[DK * 34];    // transposed [dv][key 0..31]
  __shared__ _Float16 Psh[4 * 16 * 34];

  const int tid  = threadIdx.x;
  const int lane = tid & 31;
  const int w    = tid >> 5;
  const int h    = blockIdx.y;
  const int b    = blockIdx.z;
  const int H    = gridDim.y;
  const int ld   = H * DK;
  const int q0   = blockIdx.x * 64 + w * 16;
  const int kbase = (lane >> 4) * 8;
  const float qscale = rsqrtf((float)DK);

  const float* Qb = Q  + (long long)b * Lq * ld + h * DK;
  const float* Kb = Km + (long long)b * Lk * ld + h * DK;
  const float* Vb = Vm + (long long)b * Lk * ld + h * DK;
  float*       Ob = O  + (long long)b * Lq * ld + h * DK;

  AF qf[NQF];
  {
    const int m = q0 + (lane & 15);
    const float* qr = Qb + (long long)m * ld;
    #pragma unroll
    for (int f = 0; f < NQF; ++f) {
      #pragma unroll
      for (int i = 0; i < 8; ++i) {
        int k = f * 32 + (i >> 2) * 16 + kbase + 2 * (i & 3);
        const float2 qq = *(const float2*)(qr + k);
        qf[f].u[i] = pack2(qq.x * qscale, qq.y * qscale);
      }
    }
  }

  v8f o[NOF] = {};
  float mstate[8], lstate[8];
  #pragma unroll
  for (int i = 0; i < 8; ++i) { mstate[i] = -1e30f; lstate[i] = 0.0f; }

  _Float16* Pw = &Psh[w * 16 * 34];

  for (int kb = 0; kb < Lk; kb += 32) {
    // stage K (Ksh[r][c]) and V (transposed Vsh[c][r]) with float4 loads
    for (int idx = tid; idx < 8 * DK; idx += 128) {   // 32*DK/4 quads
      const int r = idx / (DK / 4);
      const int c = (idx % (DK / 4)) * 4;
      const long long g = (long long)(kb + r) * ld + c;
      const float4 kf = *(const float4*)(Kb + g);
      *(unsigned int*)&Ksh[r * KST + c]     = pack2(kf.x, kf.y);
      *(unsigned int*)&Ksh[r * KST + c + 2] = pack2(kf.z, kf.w);
      const float4 vf = *(const float4*)(Vb + g);
      Vsh[(c + 0) * 34 + r] = (_Float16)vf.x;
      Vsh[(c + 1) * 34 + r] = (_Float16)vf.y;
      Vsh[(c + 2) * 34 + r] = (_Float16)vf.z;
      Vsh[(c + 3) * 34 + r] = (_Float16)vf.w;
    }
    __syncthreads();

    v8f s0 = {}, s1 = {};
    #pragma unroll
    for (int f = 0; f < NQF; ++f) {                    // S = Q @ K^T
      AF b0, b1;
      const int n0 = (lane & 15);
      #pragma unroll
      for (int i = 0; i < 8; ++i) {
        int k = f * 32 + (i >> 2) * 16 + kbase + 2 * (i & 3);
        b0.u[i] = *(const unsigned int*)&Ksh[n0 * KST + k];
        b1.u[i] = *(const unsigned int*)&Ksh[(n0 + 16) * KST + k];
      }
      s0 = __builtin_amdgcn_wmma_f32_16x16x32_f16(false, qf[f].v, false, b0.v, (short)0, s0, false, false);
      s1 = __builtin_amdgcn_wmma_f32_16x16x32_f16(false, qf[f].v, false, b1.v, (short)0, s1, false, false);
    }
    #pragma unroll
    for (int i = 0; i < 8; ++i) {                      // online softmax
      float v0 = s0[i], v1 = s1[i];
      float mx = fmaxf(v0, v1);
      #pragma unroll
      for (int d = 1; d < 16; d <<= 1) mx = fmaxf(mx, __shfl_xor(mx, d, 16));
      float mnew = fmaxf(mstate[i], mx);
      float p0 = __expf(v0 - mnew), p1 = __expf(v1 - mnew);
      float rs = p0 + p1;
      #pragma unroll
      for (int d = 1; d < 16; d <<= 1) rs += __shfl_xor(rs, d, 16);
      float corr = __expf(mstate[i] - mnew);
      lstate[i] = lstate[i] * corr + rs;
      mstate[i] = mnew;
      #pragma unroll
      for (int f = 0; f < NOF; ++f) o[f][i] *= corr;
      int pr = i + (lane >> 4) * 8;
      Pw[pr * 34 + (lane & 15)]      = (_Float16)p0;
      Pw[pr * 34 + 16 + (lane & 15)] = (_Float16)p1;
    }
    AF pa;                                             // P -> A-frag (LDS bounce)
    {
      const int m = (lane & 15);
      #pragma unroll
      for (int i = 0; i < 8; ++i) {
        int k = (i >> 2) * 16 + kbase + 2 * (i & 3);
        pa.u[i] = *(const unsigned int*)&Pw[m * 34 + k];
      }
    }
    #pragma unroll
    for (int f = 0; f < NOF; ++f) {                    // O += P @ V
      AF vb;
      const int n = f * 16 + (lane & 15);
      #pragma unroll
      for (int i = 0; i < 8; ++i) {
        int k = (i >> 2) * 16 + kbase + 2 * (i & 3);
        vb.u[i] = *(const unsigned int*)&Vsh[n * 34 + k];
      }
      o[f] = __builtin_amdgcn_wmma_f32_16x16x32_f16(false, pa.v, false, vb.v, (short)0, o[f], false, false);
    }
    __syncthreads();
  }
  #pragma unroll
  for (int f = 0; f < NOF; ++f) {
    const int n = f * 16 + (lane & 15);
    #pragma unroll
    for (int i = 0; i < 8; ++i) {
      const int m = q0 + i + (lane >> 4) * 8;
      Ob[(long long)m * ld + n] = o[f][i] / lstate[i];
    }
  }
}

// --------------------------- residual + LayerNorm ---------------------------
__global__ void __launch_bounds__(128) add_ln_kernel(
    const float* __restrict__ x, const float* __restrict__ r,
    const float* __restrict__ g, const float* __restrict__ beta,
    float* __restrict__ out, int D)
{
  const long long row = blockIdx.x;
  const float* xr = x + row * D;
  const float* rr = r + row * D;
  float* orow = out + row * D;
  __shared__ float red[128];
  float xv[4];
  int cnt = 0;
  float s = 0.f;
  for (int d = threadIdx.x; d < D; d += 128) { float v = xr[d] + rr[d]; xv[cnt++] = v; s += v; }
  red[threadIdx.x] = s; __syncthreads();
  for (int o = 64; o > 0; o >>= 1) { if (threadIdx.x < o) red[threadIdx.x] += red[threadIdx.x + o]; __syncthreads(); }
  float mu = red[0] / D; __syncthreads();
  s = 0.f;
  for (int i = 0; i < cnt; ++i) { float v = xv[i] - mu; s += v * v; }
  red[threadIdx.x] = s; __syncthreads();
  for (int o = 64; o > 0; o >>= 1) { if (threadIdx.x < o) red[threadIdx.x] += red[threadIdx.x + o]; __syncthreads(); }
  float inv = rsqrtf(red[0] / D + 1e-5f);
  cnt = 0;
  for (int d = threadIdx.x; d < D; d += 128) { orow[d] = (xv[cnt++] - mu) * inv * g[d] + beta[d]; }
}

// --------------------------------- misc ops ---------------------------------
__global__ void build_pts_kernel(const float* __restrict__ pin, float* __restrict__ pts,
                                 int N, int total)
{
  int t = blockIdx.x * blockDim.x + threadIdx.x;
  if (t >= total) return;
  int c = t % PTS_ST, n = (t / PTS_ST) % N, b = t / (PTS_ST * N);
  pts[t] = (c < 3 || c >= 6) ? 0.f : pin[((long long)b * 6 + c) * N + n];
}

__global__ void __launch_bounds__(256) fps_kernel(
    const float* __restrict__ xyz, int N, int stride, int S, int* __restrict__ idx)
{
  const int b = blockIdx.x;
  __shared__ float dist[NPTS];
  __shared__ float rv[256];
  __shared__ int   ri[256];
  __shared__ int   sfar;
  const int tid = threadIdx.x;
  for (int j = tid; j < N; j += 256) dist[j] = 1e10f;
  if (tid == 0) sfar = 0;
  __syncthreads();
  for (int s = 0; s < S; ++s) {
    const int far = sfar;
    if (tid == 0) idx[b * S + s] = far;
    const float cx = xyz[((long long)b * N + far) * stride + 0];
    const float cy = xyz[((long long)b * N + far) * stride + 1];
    const float cz = xyz[((long long)b * N + far) * stride + 2];
    float bv = -1.f; int bi = 0;
    for (int j = tid; j < N; j += 256) {
      float dx = xyz[((long long)b * N + j) * stride + 0] - cx;
      float dy = xyz[((long long)b * N + j) * stride + 1] - cy;
      float dz = xyz[((long long)b * N + j) * stride + 2] - cz;
      float nd = fminf(dist[j], dx * dx + dy * dy + dz * dz);
      dist[j] = nd;
      if (nd > bv) { bv = nd; bi = j; }
    }
    rv[tid] = bv; ri[tid] = bi;
    __syncthreads();
    for (int o = 128; o > 0; o >>= 1) {
      if (tid < o && (rv[tid + o] > rv[tid] || (rv[tid + o] == rv[tid] && ri[tid + o] < ri[tid]))) {
        rv[tid] = rv[tid + o]; ri[tid] = ri[tid + o];
      }
      __syncthreads();
    }
    if (tid == 0) sfar = ri[0];
    __syncthreads();
  }
}

__global__ void __launch_bounds__(256) topk_kernel(
    const float* __restrict__ score, int N, int Ksel, int* __restrict__ out)
{
  const int b = blockIdx.x;
  __shared__ float sc[NPTS];
  __shared__ float rv[256];
  __shared__ int   ri[256];
  for (int j = threadIdx.x; j < N; j += 256) sc[j] = score[(long long)b * N + j];
  __syncthreads();
  for (int s = 0; s < Ksel; ++s) {
    float bv = -1e30f; int bi = 0;
    for (int j = threadIdx.x; j < N; j += 256)
      if (sc[j] > bv) { bv = sc[j]; bi = j; }
    rv[threadIdx.x] = bv; ri[threadIdx.x] = bi;
    __syncthreads();
    for (int o = 128; o > 0; o >>= 1) {
      if (threadIdx.x < o &&
          (rv[threadIdx.x + o] > rv[threadIdx.x] ||
           (rv[threadIdx.x + o] == rv[threadIdx.x] && ri[threadIdx.x + o] < ri[threadIdx.x]))) {
        rv[threadIdx.x] = rv[threadIdx.x + o]; ri[threadIdx.x] = ri[threadIdx.x + o];
      }
      __syncthreads();
    }
    if (threadIdx.x == 0) { out[b * Ksel + s] = ri[0]; sc[ri[0]] = -1e30f; }
    __syncthreads();
  }
}

__global__ void ballquery_kernel(const float* __restrict__ xyz, int xstride,
                                 const float* __restrict__ cxyz, int N, int S, int K,
                                 float r2, int* __restrict__ gidx, int total)
{
  int t = blockIdx.x * blockDim.x + threadIdx.x;
  if (t >= total) return;
  int b = t / S;
  const float cx = cxyz[t * 3 + 0], cy = cxyz[t * 3 + 1], cz = cxyz[t * 3 + 2];
  int* out = gidx + (long long)t * K;
  int cnt = 0, first = -1;
  const float* base = xyz + (long long)b * N * xstride;
  for (int j = 0; j < N; ++j) {
    float dx = base[j * xstride + 0] - cx;
    float dy = base[j * xstride + 1] - cy;
    float dz = base[j * xstride + 2] - cz;
    if (dx * dx + dy * dy + dz * dz <= r2) {
      if (first < 0) first = j;
      if (cnt < K) out[cnt++] = j; else break;
    }
  }
  if (first < 0) first = N - 1;
  for (; cnt < K; ++cnt) out[cnt] = first;
}

__global__ void __launch_bounds__(128) group_kernel(
    const float* __restrict__ feats, int D, const float* __restrict__ xyz, int xstride,
    const float* __restrict__ cxyz, const int* __restrict__ gidx,
    int N, int S, int K, int Dtp, float* __restrict__ out)
{
  const int blk = blockIdx.x;           // b*S*K + s*K + k
  const int s = (blk / K) % S;
  const int b = blk / (K * S);
  const int j = gidx[blk];
  const int Dt = D + 3;
  const float* f  = feats + ((long long)b * N + j) * D;
  const float* xr = xyz   + ((long long)b * N + j) * xstride;
  const float* cr = cxyz  + ((long long)b * S + s) * 3;
  float* orow = out + (long long)blk * Dtp;
  for (int c = threadIdx.x; c < Dtp; c += 128)
    orow[c] = (c < D) ? f[c] : ((c < Dt) ? (xr[c - D] - cr[c - D]) : 0.0f);
}

__global__ void __launch_bounds__(128) maxpool_kernel(
    const float* __restrict__ in, int Din, int K,
    float* __restrict__ out, int ldo, long long so, int coff, int S)
{
  const int s = blockIdx.x, b = blockIdx.y;
  const float* base = in + (long long)(b * S + s) * K * Din;
  float* orow = out + (long long)b * so + (long long)s * ldo + coff;
  for (int d = threadIdx.x; d < Din; d += 128) {
    float m = -1e30f;
    for (int k = 0; k < K; ++k) m = fmaxf(m, base[(long long)k * Din + d]);
    orow[d] = m;
  }
}

__global__ void gather_kernel(float* __restrict__ dst, long long dBS, int dStride,
                              const float* __restrict__ src, long long sBS, int sStride,
                              const int* __restrict__ idx, int idxStride,
                              int S, int len, int total)
{
  int t = blockIdx.x * blockDim.x + threadIdx.x;
  if (t >= total) return;
  int c = t % len, s = (t / len) % S, b = t / (len * S);
  dst[b * dBS + (long long)s * dStride + c] =
      src[b * sBS + (long long)idx[(long long)b * idxStride + s] * sStride + c];
}

// ============================ host orchestration ============================

static void gemm(hipStream_t st, const float* A, const float* W, const float* bias,
                 float* C, int batch, int M, int N, int K,
                 int lda, long long sA, int ldc, long long sC, int act)
{
  dim3 g((N + 63) / 64, (M + 63) / 64, batch);
  gemm_kernel<<<g, dim3(128), 0, st>>>(A, W, bias, C, M, N, K, lda, sA, ldc, sC, act);
}

struct MhaP { const float *wq,*bq,*wk,*bk,*wv,*bv,*wo,*bo,*g,*be; };

static void run_mha(hipStream_t st, const MhaP& p, const float* xq, const float* xkv,
                    int Lq, int Lk, int Dq, int Dkv, int H, int dk,
                    float* out, float* arena)
{
  const int Dh = H * dk;
  float* Qb = arena;
  float* Kb = Qb + (size_t)BB * Lq * Dh;
  float* Vb = Kb + (size_t)BB * Lk * Dh;
  float* Ab = Vb + (size_t)BB * Lk * Dh;
  float* Pb = Ab + (size_t)BB * Lq * Dh;
  gemm(st, xq,  p.wq, p.bq, Qb, BB, Lq, Dh, Dq,  Dq,  (long long)Lq * Dq,  Dh, (long long)Lq * Dh, 0);
  gemm(st, xkv, p.wk, p.bk, Kb, BB, Lk, Dh, Dkv, Dkv, (long long)Lk * Dkv, Dh, (long long)Lk * Dh, 0);
  gemm(st, xkv, p.wv, p.bv, Vb, BB, Lk, Dh, Dkv, Dkv, (long long)Lk * Dkv, Dh, (long long)Lk * Dh, 0);
  dim3 g(Lq / 64, H, BB);
  if (dk == 128) flash_attn_kernel<128><<<g, dim3(128), 0, st>>>(Qb, Kb, Vb, Ab, Lq, Lk);
  else           flash_attn_kernel<64> <<<g, dim3(128), 0, st>>>(Qb, Kb, Vb, Ab, Lq, Lk);
  gemm(st, Ab, p.wo, p.bo, Pb, BB, Lq, Dq, Dh, Dh, (long long)Lq * Dh, Dq, (long long)Lq * Dq, 0);
  add_ln_kernel<<<dim3(BB * Lq), dim3(128), 0, st>>>(Pb, xq, p.g, p.be, out, Dq);
}

static void run_sa(hipStream_t st, const float* const* L, int saBase,
                   const float* feats, int D, const float* xyz,
                   const int* seed, int S, float* fa, float* nxyz, int* gidx, float* arena)
{
  int tot = BB * S * 3;
  gather_kernel<<<(tot + 255) / 256, 256, 0, st>>>(
      nxyz, (long long)S * 3, 3, xyz, (long long)NPTS * PTS_ST, PTS_ST, seed, S, S, 3, tot);
  const float radii[3] = {0.1f, 0.2f, 0.4f};
  const int   ns[3]    = {16, 32, 128};
  const int   mlps[3][3] = {{32, 32, 64}, {64, 64, 128}, {64, 96, 128}};
  const int   coffs[3] = {0, 64, 192};
  const int Dt  = D + 3;
  const int Dtp = (Dt + 3) & ~3;      // padded grouped row stride (516 / 260)
  int leaf = saBase;
  for (int br = 0; br < 3; ++br) {
    const int K = ns[br];
    const int Mrows = S * K;
    float* grouped = arena;
    float* ping = grouped + (size_t)BB * Mrows * Dtp;
    float* pong = ping + (size_t)BB * Mrows * 128;
    const int tcnt = BB * S;
    ballquery_kernel<<<(tcnt + 127) / 128, 128, 0, st>>>(
        xyz, PTS_ST, nxyz, NPTS, S, K, radii[br] * radii[br], gidx, tcnt);
    group_kernel<<<dim3(BB * Mrows), dim3(128), 0, st>>>(
        feats, D, xyz, PTS_ST, nxyz, gidx, NPTS, S, K, Dtp, grouped);
    const int n0 = mlps[br][0], n1 = mlps[br][1], n2 = mlps[br][2];
    gemm(st, grouped, L[leaf + 0], L[leaf + 1], ping, BB, Mrows, n0, Dt, Dtp, (long long)Mrows * Dtp, n0, (long long)Mrows * n0, 1);
    gemm(st, ping,    L[leaf + 2], L[leaf + 3], pong, BB, Mrows, n1, n0, n0,  (long long)Mrows * n0,  n1, (long long)Mrows * n1, 1);
    gemm(st, pong,    L[leaf + 4], L[leaf + 5], ping, BB, Mrows, n2, n1, n1,  (long long)Mrows * n1,  n2, (long long)Mrows * n2, 1);
    maxpool_kernel<<<dim3(S, BB), dim3(128), 0, st>>>(
        ping, n2, K, fa, 320, (long long)S * 320, coffs[br], S);
    leaf += 6;
  }
}

// Assumed leaf order: d_in[0]=points, then params flattened in JAX tree order
// (dict keys sorted alphabetically, lists/tuples positional):
//   fc(6) fc_cls(8) gfg{att(fc,ln,wk,wq,wv)=10, fc=6, fc_agg=6, sa=18}
//   lfg{att=10, fc=6, sortnets 4x[fc=6, fc_agg=6, sa=18]} lg_att(10)
enum {
  LI_FC = 1, LI_FCCLS = 7,
  LI_GFG_ATT = 15, LI_GFG_FC = 25, LI_GFG_AGG = 31, LI_GFG_SA = 37,
  LI_LFG_ATT = 55, LI_LFG_FC = 65, LI_SN0 = 71, LI_LG_ATT = 191,
};

extern "C" void kernel_launch(void* const* d_in, const int* in_sizes, int n_in,
                              void* d_out, int out_size, void* d_ws, size_t ws_size,
                              hipStream_t stream)
{
  (void)in_sizes; (void)out_size; (void)ws_size;
  const float* L[201];
  for (int i = 0; i < 201 && i < n_in; ++i) L[i] = (const float*)d_in[i];
  const float* points = L[0];

  float* W = (float*)d_ws;
  size_t off = 0;
  auto alloc = [&](size_t n) { float* p = W + off; off += (n + 31) & ~(size_t)31; return p; };
  float* pts   = alloc((size_t)BB * NPTS * PTS_ST);
  float* xl    = alloc((size_t)BB * NPTS * 512);
  float* xg    = alloc((size_t)BB * NPTS * 256);
  float* local = alloc((size_t)BB * 256 * 512);
  float* glob  = alloc((size_t)BB * 128 * 256);
  float* fa    = alloc((size_t)BB * 128 * 320);
  float* sc    = alloc((size_t)BB * NPTS);
  float* nxyz  = alloc((size_t)BB * 128 * 3);
  int*   seed  = (int*)alloc((size_t)BB * 128);
  int*   gidx  = (int*)alloc((size_t)BB * 128 * 128);
  float* lg    = alloc((size_t)BB * 256 * 512);
  float* arena = W + off;   // reused phase scratch (peak ~34M floats)

  auto mhaP = [&](int base) {
    MhaP p;
    p.wo = L[base + 0]; p.bo = L[base + 1];
    p.g  = L[base + 2]; p.be = L[base + 3];
    p.wk = L[base + 4]; p.bk = L[base + 5];
    p.wq = L[base + 6]; p.bq = L[base + 7];
    p.wv = L[base + 8]; p.bv = L[base + 9];
    return p;
  };

  // pts = transpose(concat(zeros(3), points[3:]))  -> (B, N, 8), cols 6..7 = 0
  const int totp = BB * NPTS * PTS_ST;
  build_pts_kernel<<<(totp + 255) / 256, 256, 0, stream>>>(points, pts, NPTS, totp);

  // ---- local feature generation (lfg) ----
  {
    float* h1 = arena;
    float* h2 = h1 + (size_t)BB * NPTS * 64;
    gemm(stream, pts, L[LI_LFG_FC + 0], L[LI_LFG_FC + 1], h1, BB, NPTS, 64, 6, PTS_ST, (long long)NPTS * PTS_ST, 64, (long long)NPTS * 64, 1);
    gemm(stream, h1,  L[LI_LFG_FC + 2], L[LI_LFG_FC + 3], h2, BB, NPTS, 256, 64, 64, (long long)NPTS * 64, 256, (long long)NPTS * 256, 1);
    gemm(stream, h2,  L[LI_LFG_FC + 4], L[LI_LFG_FC + 5], xl, BB, NPTS, 512, 256, 256, (long long)NPTS * 256, 512, (long long)NPTS * 512, 0);
  }
  run_mha(stream, mhaP(LI_LFG_ATT), xl, xl, NPTS, NPTS, 512, 512, 4, 128, xl, arena);

  // ---- 4 sortnets -> local (B, 256, 512) ----
  for (int sn = 0; sn < 4; ++sn) {
    const int base = LI_SN0 + sn * 30;
    float* a1 = arena;
    float* a2 = a1 + (size_t)BB * NPTS * 256;
    float* saArena = a2 + (size_t)BB * NPTS * 64;
    gemm(stream, xl, L[base + 0], L[base + 1], a1, BB, NPTS, 256, 512, 512, (long long)NPTS * 512, 256, (long long)NPTS * 256, 1);
    gemm(stream, a1, L[base + 2], L[base + 3], a2, BB, NPTS, 64, 256, 256, (long long)NPTS * 256, 64, (long long)NPTS * 64, 1);
    gemm(stream, a2, L[base + 4], L[base + 5], sc, BB, NPTS, 1, 64, 64, (long long)NPTS * 64, 1, (long long)NPTS, 0);
    topk_kernel<<<BB, 256, 0, stream>>>(sc, NPTS, 64, seed);
    run_sa(stream, L, base + 12, xl, 512, pts, seed, 64, fa, nxyz, gidx, saArena);
    float* g1 = saArena;
    float* g2 = g1 + (size_t)BB * 64 * 256;
    gemm(stream, fa, L[base + 6], L[base + 7], g1, BB, 64, 256, 320, 320, (long long)64 * 320, 256, (long long)64 * 256, 1);
    gemm(stream, g1, L[base + 8], L[base + 9], g2, BB, 64, 256, 256, 256, (long long)64 * 256, 256, (long long)64 * 256, 1);
    gemm(stream, g2, L[base + 10], L[base + 11], local + (size_t)sn * 64 * 512,
         BB, 64, 505, 256, 256, (long long)64 * 256, 512, (long long)256 * 512, 0);
    int t1 = BB * 64 * 1;
    gather_kernel<<<(t1 + 255) / 256, 256, 0, stream>>>(
        local + (size_t)sn * 64 * 512 + 505, (long long)256 * 512, 512,
        sc, (long long)NPTS, 1, seed, 64, 64, 1, t1);
    int t6 = BB * 64 * 6;
    gather_kernel<<<(t6 + 255) / 256, 256, 0, stream>>>(
        local + (size_t)sn * 64 * 512 + 506, (long long)256 * 512, 512,
        pts, (long long)NPTS * PTS_ST, PTS_ST, seed, 64, 64, 6, t6);
  }

  // ---- global feature generation (gfg) ----
  {
    float* h1 = arena;
    float* h2 = h1 + (size_t)BB * NPTS * 64;
    gemm(stream, pts, L[LI_GFG_FC + 0], L[LI_GFG_FC + 1], h1, BB, NPTS, 64, 6, PTS_ST, (long long)NPTS * PTS_ST, 64, (long long)NPTS * 64, 1);
    gemm(stream, h1,  L[LI_GFG_FC + 2], L[LI_GFG_FC + 3], h2, BB, NPTS, 256, 64, 64, (long long)NPTS * 64, 256, (long long)NPTS * 256, 1);
    gemm(stream, h2,  L[LI_GFG_FC + 4], L[LI_GFG_FC + 5], xg, BB, NPTS, 256, 256, 256, (long long)NPTS * 256, 256, (long long)NPTS * 256, 0);
  }
  run_mha(stream, mhaP(LI_GFG_ATT), xg, xg, NPTS, NPTS, 256, 256, 4, 64, xg, arena);
  fps_kernel<<<BB, 256, 0, stream>>>(pts, NPTS, PTS_ST, 128, seed);
  run_sa(stream, L, LI_GFG_SA, xg, 256, pts, seed, 128, fa, nxyz, gidx, arena);
  {
    float* g1 = arena;
    float* g2 = g1 + (size_t)BB * 128 * 256;
    gemm(stream, fa, L[LI_GFG_AGG + 0], L[LI_GFG_AGG + 1], g1, BB, 128, 256, 320, 320, (long long)128 * 320, 256, (long long)128 * 256, 1);
    gemm(stream, g1, L[LI_GFG_AGG + 2], L[LI_GFG_AGG + 3], g2, BB, 128, 256, 256, 256, (long long)128 * 256, 256, (long long)128 * 256, 1);
    gemm(stream, g2, L[LI_GFG_AGG + 4], L[LI_GFG_AGG + 5], glob, BB, 128, 256, 256, 256, (long long)128 * 256, 256, (long long)128 * 256, 0);
  }

  // ---- local-global cross attention + head ----
  run_mha(stream, mhaP(LI_LG_ATT), local, glob, 256, 128, 512, 256, 8, 64, lg, arena);
  {
    float* c1 = arena;
    float* c2 = c1 + (size_t)BB * 256 * 256;
    float* c3 = c2 + (size_t)BB * 256 * 256;
    gemm(stream, lg, L[LI_FC + 0], L[LI_FC + 1], c1, BB, 256, 256, 512, 512, (long long)256 * 512, 256, (long long)256 * 256, 1);
    gemm(stream, c1, L[LI_FC + 2], L[LI_FC + 3], c2, BB, 256, 256, 256, 256, (long long)256 * 256, 256, (long long)256 * 256, 1);
    gemm(stream, c2, L[LI_FC + 4], L[LI_FC + 5], c3, BB, 256, 64, 256, 256, (long long)256 * 256, 64, (long long)256 * 64, 0);
    // classifier: (4, 16384) rows, single "batch"
    float* d1 = c3 + (size_t)BB * 256 * 64;
    float* d2 = d1 + (size_t)BB * 1024;
    float* d3 = d2 + (size_t)BB * 256;
    gemm(stream, c3, L[LI_FCCLS + 0], L[LI_FCCLS + 1], d1, 1, BB, 1024, 16384, 16384, 0, 1024, 0, 0);
    gemm(stream, d1, L[LI_FCCLS + 2], L[LI_FCCLS + 3], d2, 1, BB, 256, 1024, 1024, 0, 256, 0, 1);
    gemm(stream, d2, L[LI_FCCLS + 4], L[LI_FCCLS + 5], d3, 1, BB, 64, 256, 256, 0, 64, 0, 1);
    gemm(stream, d3, L[LI_FCCLS + 6], L[LI_FCCLS + 7], (float*)d_out, 1, BB, 40, 64, 64, 0, 40, 0, 0);
  }
}
